// CustomLSTMLayer_72335839199509
// MI455X (gfx1250) — compile-verified, matrix operand order checked by reading, not run on hardware
//
#include <hip/hip_runtime.h>

#define S_LEN 512
#define BATCH 64
#define DIM   1024
#define HS    1024
#define NG    4096           // 4*HS
#define KT_DIM (DIM / 32)    // 32 k-tiles
#define NT_DIM (NG / 16)     // 256 n-tiles

typedef __attribute__((ext_vector_type(16))) __bf16 v16bf;
typedef __attribute__((ext_vector_type(8)))  float  v8f;

// ---------- bf16 helpers (native cvt, RNE) ----------
__device__ __forceinline__ float bf16q(float f) {            // fake-quant f32->bf16->f32
    return (float)(__bf16)f;
}
__device__ __forceinline__ float bf16_to_f32(__bf16 h) {
    unsigned u = (unsigned)__builtin_bit_cast(unsigned short, h) << 16;
    return __builtin_bit_cast(float, u);
}

// ---------- BFP(block=8, MAN=7) quantize 8 contiguous values -> bf16 (lossless) ----
// maxabs = m*2^e, m in [0.5,1)  =>  step = 2^(e-8); q = rint(x/step)*step, |q/step|<=256
// every such q is exactly representable in bf16.  maxabs==0 => all x==0 => q==0, so
// no select needed (denormal/inf edge cases don't occur in this workload).
__device__ __forceinline__ void quant8_bf16(const float* x, __bf16* o) {
    float m = fabsf(x[0]);
#pragma unroll
    for (int i = 1; i < 8; ++i) m = fmaxf(m, fabsf(x[i]));
    unsigned mb = __builtin_bit_cast(unsigned, m);
    int be = (int)(mb >> 23);                 // biased exponent of maxabs
    float step = __builtin_bit_cast(float, ((unsigned)(be - 7)  & 0xFFu) << 23);
    float inv  = __builtin_bit_cast(float, ((unsigned)(261 - be) & 0xFFu) << 23);
#pragma unroll
    for (int i = 0; i < 8; ++i)
        o[i] = (__bf16)(__builtin_rintf(x[i] * inv) * step);
}

__device__ __forceinline__ void load8(const float* p, float* x) {
    const float4 a = *(const float4*)(p);
    const float4 b = *(const float4*)(p + 4);
    x[0] = a.x; x[1] = a.y; x[2] = a.z; x[3] = a.w;
    x[4] = b.x; x[5] = b.y; x[6] = b.z; x[7] = b.w;
}

// ---------- fragment loaders ----------
template<int NB, int NSTRIDE>
__device__ __forceinline__ void load_bfrag(const __bf16* __restrict__ Bp, int KTiles,
                                           int nt0, int kt, int lane, v16bf* dst) {
#pragma unroll
    for (int j = 0; j < NB; ++j) {
        const __bf16* bt = Bp + ((size_t)(nt0 + j * NSTRIDE) * KTiles + kt) * 512 + lane * 16;
        __builtin_prefetch(bt + 1024, 0, 1);             // 2 k-tiles ahead
        dst[j] = *(const v16bf*)bt;
    }
}

__device__ __forceinline__ v16bf load_quant_a(const float* __restrict__ Arow,
                                              int kt, int abase) {
    const int k0 = kt * 32 + abase;
    __builtin_prefetch(Arow + k0 + 64, 0, 1);            // 2 k-tiles ahead
    float x[16];
    load8(Arow + k0,      x);
    load8(Arow + k0 + 16, x + 8);
    v16bf a;
    quant8_bf16(x,     (__bf16*)&a);
    quant8_bf16(x + 8, ((__bf16*)&a) + 8);
    return a;
}

// ---------- software-pipelined K-loop, unrolled x2 with alternating B buffers ------
// No buffer rotation copies: b0/b1 swap roles each half-step, so the loads for one
// buffer issue while the other buffer feeds 4 back-to-back WMMAs (BFP quant VALU
// co-executes with the XDL WMMA pipe).  KTiles must be even (it is 32 here).
template<int NB, int NSTRIDE>
__device__ __forceinline__ void gemm_core(const float* __restrict__ Arow,
                                          const __bf16* __restrict__ Bp,
                                          int KTiles, int lane, int nt0, v8f* acc) {
    const int abase = (lane & 16) ? 8 : 0;
    v16bf b0[NB], b1[NB];
    load_bfrag<NB, NSTRIDE>(Bp, KTiles, nt0, 0, lane, b0);
    for (int kt = 0; kt < KTiles; kt += 2) {
        load_bfrag<NB, NSTRIDE>(Bp, KTiles, nt0, kt + 1, lane, b1);
        v16bf a0 = load_quant_a(Arow, kt, abase);
#pragma unroll
        for (int j = 0; j < NB; ++j)
            acc[j] = __builtin_amdgcn_wmma_f32_16x16x32_bf16(
                false, a0, false, b0[j], (short)0, acc[j], false, false);
        const int kt2 = (kt + 2 < KTiles) ? kt + 2 : kt;  // clamped, uniform
        load_bfrag<NB, NSTRIDE>(Bp, KTiles, nt0, kt2, lane, b0);
        v16bf a1 = load_quant_a(Arow, kt + 1, abase);
#pragma unroll
        for (int j = 0; j < NB; ++j)
            acc[j] = __builtin_amdgcn_wmma_f32_16x16x32_bf16(
                false, a1, false, b1[j], (short)0, acc[j], false, false);
    }
}

// ---------- pack W [4096, K] row-major f32 -> BFP-bf16 WMMA-B tiles ----------------
__global__ void pack_weights_kernel(const float* __restrict__ W, __bf16* __restrict__ Bp, int K) {
    const int tid  = blockIdx.x * blockDim.x + threadIdx.x;   // one thread per (tile,lane)
    const int lane = tid & 31;
    const int tile = tid >> 5;
    const int KTt  = K >> 5;
    const int nt = tile / KTt, kt = tile % KTt;
    const int n  = nt * 16 + (lane & 15);
    const int k0 = kt * 32 + ((lane & 16) ? 8 : 0);
    const float* Wr = W + (size_t)n * K;
    float x[16];
    load8(Wr + k0,      x);
    load8(Wr + k0 + 16, x + 8);
    v16bf vv;
    quant8_bf16(x,     (__bf16*)&vv);
    quant8_bf16(x + 8, ((__bf16*)&vv) + 8);
    *(v16bf*)(Bp + ((size_t)tile * 32 + lane) * 16) = vv;
}

__global__ void make_bias_kernel(const float* __restrict__ bih, const float* __restrict__ bhh,
                                 float* __restrict__ bias) {
    int i = blockIdx.x * blockDim.x + threadIdx.x;
    bias[i] = bf16q(bih[i]) + bf16q(bhh[i]);
}

// ---------- phase 1: Xg = bf16(x_q @ U^T_q + b_q), one 32768x4096x1024 GEMM --------
__global__ void __launch_bounds__(256, 1)
gemm_xU_kernel(const float* __restrict__ X, const __bf16* __restrict__ Up,
               const float* __restrict__ bias, __bf16* __restrict__ Xg) {
    constexpr int NB = 4;
    const int lane = threadIdx.x & 31;
    const size_t wid = ((size_t)blockIdx.x * blockDim.x + threadIdx.x) >> 5;
    const int mt  = (int)(wid / (NT_DIM / NB));          // 0..2047
    const int nt0 = (int)(wid % (NT_DIM / NB)) * NB;     // adjacent n-tiles
    const int lrow = lane & 15;
    const float* Arow = X + (size_t)(mt * 16 + lrow) * DIM;
    v8f acc[NB];
#pragma unroll
    for (int j = 0; j < NB; ++j) {
        float bv = bias[(nt0 + j) * 16 + lrow];
#pragma unroll
        for (int v = 0; v < 8; ++v) acc[j][v] = bv;
    }
    gemm_core<NB, 1>(Arow, Up, KT_DIM, lane, nt0, acc);
    const int rbase = mt * 16 + ((lane >> 4) << 3);      // C layout: VGPR v -> row v / v+8
#pragma unroll
    for (int j = 0; j < NB; ++j) {
        const int col = (nt0 + j) * 16 + lrow;
#pragma unroll
        for (int v = 0; v < 8; ++v)
            Xg[(size_t)(rbase + v) * NG + col] = (__bf16)acc[j][v];
    }
}

// ---------- phase 2: fused  h@V^T GEMM + LSTM cell, one kernel per timestep --------
// A wave owns the SAME 16-column slab of all four gates (tiles nt0, nt0+64, nt0+128,
// nt0+192): after the K-loop it holds matching i/f/g/o fragments and applies the
// nonlinearities + c/h update in registers.  No gates buffer, 1 launch per step.
__global__ void __launch_bounds__(256, 1)
lstm_step_kernel(const float* __restrict__ H, const __bf16* __restrict__ Vp,
                 const __bf16* __restrict__ Xg_s, float* __restrict__ c,
                 float* __restrict__ h_out) {
    const int lane = threadIdx.x & 31;
    const int wid  = (blockIdx.x * blockDim.x + threadIdx.x) >> 5;   // 0..255
    const int mt  = wid >> 6;                            // 0..3
    const int nt0 = wid & 63;                            // 16-col slab within HS
    const int lrow = lane & 15;
    const float* Arow = H + (size_t)(mt * 16 + lrow) * HS;
    const int rbase = mt * 16 + ((lane >> 4) << 3);
    v8f acc[4];
#pragma unroll
    for (int g = 0; g < 4; ++g) {                        // C-init from precomputed x@U+b
        const int col = g * HS + nt0 * 16 + lrow;
#pragma unroll
        for (int v = 0; v < 8; ++v)
            acc[g][v] = bf16_to_f32(Xg_s[(size_t)(rbase + v) * NG + col]);
    }
    gemm_core<4, 64>(Arow, Vp, HS / 32, lane, nt0, acc);

    const int col = nt0 * 16 + lrow;
#pragma unroll
    for (int v = 0; v < 8; ++v) {
        const size_t idx = (size_t)(rbase + v) * HS + col;
        float i_t = bf16q(1.0f / (1.0f + __expf(-bf16q(acc[0][v]))));
        float f_t = bf16q(1.0f / (1.0f + __expf(-bf16q(acc[1][v]))));
        float g_t = bf16q(tanhf(bf16q(acc[2][v])));
        float o_t = bf16q(1.0f / (1.0f + __expf(-bf16q(acc[3][v]))));
        float cn  = bf16q(f_t * c[idx] + i_t * g_t);
        c[idx] = cn;
        h_out[idx] = bf16q(o_t * tanhf(cn));
    }
}

extern "C" void kernel_launch(void* const* d_in, const int* in_sizes, int n_in,
                              void* d_out, int out_size, void* d_ws, size_t ws_size,
                              hipStream_t stream) {
    const float* X   = (const float*)d_in[0];   // [512,64,1024]
    const float* h0  = (const float*)d_in[1];   // [64,1024]
    const float* c0  = (const float*)d_in[2];   // [64,1024]
    const float* U   = (const float*)d_in[3];   // [4096,1024]
    const float* V   = (const float*)d_in[4];   // [4096,1024]
    const float* bih = (const float*)d_in[5];   // [4096]
    const float* bhh = (const float*)d_in[6];   // [4096]
    float* out = (float*)d_out;                 // [512*64*1024] ++ h_T ++ c_T

    char* ws = (char*)d_ws;
    __bf16* Up   = (__bf16*)(ws);                          //  8 MB packed U^T tiles
    __bf16* Vp   = (__bf16*)(ws + (size_t)(8u  << 20));    //  8 MB packed V^T tiles
    float*  bias = (float*) (ws + (size_t)(16u << 20));    // 16 KB
    float*  cbuf = (float*) (ws + (size_t)(17u << 20));    // 256 KB cell state
    __bf16* Xg   = (__bf16*)(ws + (size_t)(18u << 20));    // 256 MB precomputed x@U+b

    const size_t BH = (size_t)BATCH * HS;                  // 65536

    // one-time weight quantize + pack into WMMA-B lane layout
    pack_weights_kernel<<<1024, 256, 0, stream>>>(U, Up, DIM);
    pack_weights_kernel<<<1024, 256, 0, stream>>>(V, Vp, HS);
    make_bias_kernel<<<NG / 256, 256, 0, stream>>>(bih, bhh, bias);

    // phase 1: full-occupancy GEMM over all 512 timesteps at once
    // waves = (32768/16) * (256/4) = 131072  ->  16384 blocks x 8 waves
    gemm_xU_kernel<<<16384, 256, 0, stream>>>(X, Up, bias, Xg);

    hipMemcpyAsync(cbuf, c0, BH * sizeof(float), hipMemcpyDeviceToDevice, stream);

    // phase 2: sequential recurrence, 1 fused kernel per step (V is L2-resident)
    for (int s = 0; s < S_LEN; ++s) {
        const float* hprev = (s == 0) ? h0 : out + (size_t)(s - 1) * BH;
        lstm_step_kernel<<<32, 256, 0, stream>>>(hprev, Vp, Xg + (size_t)s * BATCH * NG,
                                                 cbuf, out + (size_t)s * BH);
    }

    // tuple tail: (h_T, c_T)
    hipMemcpyAsync(out + (size_t)S_LEN * BH, out + (size_t)(S_LEN - 1) * BH,
                   BH * sizeof(float), hipMemcpyDeviceToDevice, stream);
    hipMemcpyAsync(out + (size_t)S_LEN * BH + BH, cbuf,
                   BH * sizeof(float), hipMemcpyDeviceToDevice, stream);
}
